// Block_17884243820768
// MI455X (gfx1250) — matrix operand order repaired and checked by reading, not compiled, not run on hardware
//
#include <hip/hip_runtime.h>
#include <hip/hip_bf16.h>
#include <math.h>

// ---------------------------------------------------------------------------
// Problem constants (match reference)
// ---------------------------------------------------------------------------
#define BB   2
#define TT   2048
#define CC   256
#define HH   16
#define DD   16
#define BT   (BB * TT)          // 4096 tokens
#define BTC  (BB * TT * CC)     // 1,048,576 elements
#define MLPH 1024

typedef _Float16 half_t;
typedef __attribute__((ext_vector_type(16))) _Float16 v16h;
typedef __attribute__((ext_vector_type(8)))  _Float16 v8h;
typedef __attribute__((ext_vector_type(8)))  float    v8f;

#define WMMA_F16(A, Bm, Cacc) \
  __builtin_amdgcn_wmma_f32_16x16x32_f16(false, (A), false, (Bm), (short)0, (Cacc), false, false)

static __device__ __forceinline__ v16h cat8(v8h lo, v8h hi) {
  return __builtin_shufflevector(lo, hi, 0,1,2,3,4,5,6,7,8,9,10,11,12,13,14,15);
}

// Async copy of 16 bytes global -> LDS (per lane), tracked by ASYNCcnt.
static __device__ __forceinline__ void async_b128(const half_t* g, void* lds) {
  unsigned ldsoff = (unsigned)(uintptr_t)lds;   // flat LDS addressing uses addr[31:0]
  asm volatile("global_load_async_to_lds_b128 %0, %1, off"
               :: "v"(ldsoff), "v"(g) : "memory");
}
static __device__ __forceinline__ void wait_async_le4() {
  asm volatile("s_wait_asynccnt 0x4" ::: "memory");
}
static __device__ __forceinline__ void wait_async_0() {
  asm volatile("s_wait_asynccnt 0x0" ::: "memory");
}

// ---------------------------------------------------------------------------
// f32 -> f16 conversion kernel (n multiple of 4)
// ---------------------------------------------------------------------------
__global__ __launch_bounds__(256) void cvt_kernel(
    const float* __restrict__ s, half_t* __restrict__ d, int n)
{
  int i = (blockIdx.x * 256 + threadIdx.x) * 4;
  if (i < n) {
    float4 v = *reinterpret_cast<const float4*>(s + i);
    d[i + 0] = (half_t)v.x;
    d[i + 1] = (half_t)v.y;
    d[i + 2] = (half_t)v.z;
    d[i + 3] = (half_t)v.w;
  }
}

// ---------------------------------------------------------------------------
// GEMM:  Y[M,N] = act(Xh[M,K] @ Wh[K,N] + bias[N]) (+ R[M,N])
// f16 inputs, f32 output, f16 WMMA with f32 accumulation.
// 64x64 block tile, BK=32, 4 waves; double-buffered LDS fed by async-to-LDS.
// ---------------------------------------------------------------------------
__global__ __launch_bounds__(128) void gemm16_kernel(
    const half_t* __restrict__ Xh, const half_t* __restrict__ Wh,
    const float* __restrict__ bias, const float* __restrict__ R,
    float* __restrict__ Y, int M, int N, int K, int act)
{
  constexpr int BM = 64, BN = 64, BK = 32;
  constexpr int AS = 40;   // padded LDS row stride (f16) for A, 80B (16B aligned)
  constexpr int BS = 72;   // padded LDS row stride (f16) for B, 144B (16B aligned)
  __shared__ half_t As[2][BM * AS];
  __shared__ half_t Bs[2][BK * BS];

  const int tid  = threadIdx.x;
  const int lane = tid & 31;
  const int wave = tid >> 5;
  const int m0   = blockIdx.y * BM;
  const int n0   = blockIdx.x * BN;

  const int mrow = lane & 15;
  const int kofs = (lane >> 4) * 8;

  // Each thread moves 2 16B chunks of A (64x32 f16) and 2 of B (32x64 f16).
  auto issue = [&](int buf, int k0) {
#pragma unroll
    for (int i = 0; i < 2; ++i) {
      int ch = tid + i * 128;                 // A: 256 chunks, 4 per row
      int ra = ch >> 2, ca = (ch & 3) * 8;
      async_b128(Xh + (size_t)(m0 + ra) * K + k0 + ca, &As[buf][ra * AS + ca]);
      int rb = ch >> 3, cb = (ch & 7) * 8;    // B: 256 chunks, 8 per row
      async_b128(Wh + (size_t)(k0 + rb) * N + n0 + cb, &Bs[buf][rb * BS + cb]);
    }
  };

  v8f acc[4] = {};
  issue(0, 0);                                 // prologue: stage 0
  int cur = 0;
  for (int k0 = 0; k0 < K; k0 += BK) {
    if (k0 + BK < K) {                         // overlap next stage with wait
      issue(cur ^ 1, k0 + BK);
      wait_async_le4();                        // stage k complete (in-order)
    } else {
      wait_async_0();
    }
    __syncthreads();                           // cross-wave visibility

    // A operand: lane = row (wave*16 + mrow), K split per ISA layout
    const half_t* ap = &As[cur][(wave * 16 + mrow) * AS];
    v8h alo = *reinterpret_cast<const v8h*>(ap + kofs);        // K=kofs..kofs+7
    v8h ahi = *reinterpret_cast<const v8h*>(ap + kofs + 16);   // K=kofs+16..+23
    v16h aop = cat8(alo, ahi);

#pragma unroll
    for (int t = 0; t < 4; ++t) {
      // B operand: lane = N column (mrow), 16 K values in table order
      v16h bop;
#pragma unroll
      for (int s = 0; s < 16; ++s)
        bop[s] = Bs[cur][(kofs + (s < 8 ? s : s + 8)) * BS + t * 16 + mrow];
      acc[t] = WMMA_F16(aop, bop, acc[t]);
    }
    __syncthreads();                           // all reads done before buffer reuse
    cur ^= 1;
  }

  // ---- epilogue: bias, activation, residual ----
  const int rowbase = m0 + wave * 16 + ((lane >> 4) ? 8 : 0);
  const int colbase = n0 + (lane & 15);
#pragma unroll
  for (int t = 0; t < 4; ++t) {
#pragma unroll
    for (int r = 0; r < 8; ++r) {
      int row = rowbase + r;
      int col = colbase + t * 16;
      float v = acc[t][r] + bias[col];
      if (act == 1) v = 0.5f * v * (1.0f + erff(v * 0.70710678118654752f)); // exact GELU
      if (R) v += R[(size_t)row * N + col];
      Y[(size_t)row * N + col] = v;
    }
  }
}

// ---------------------------------------------------------------------------
// Flash attention: one wave per (b, h, 16-row q tile). D=16 padded to K=32.
// 32 keys per iteration: 2 score WMMAs + online softmax + 1 PV WMMA.
// ---------------------------------------------------------------------------
__global__ __launch_bounds__(32) void flash_kernel(
    const float* __restrict__ Q, const float* __restrict__ Kin,
    const float* __restrict__ V, float* __restrict__ Y)
{
  __shared__ half_t Ks[32 * 16];   // 32 keys x 16 dims
  __shared__ half_t Ps[16 * 32];   // 16 rows x 32 keys

  const int lane = threadIdx.x;
  const int q0   = blockIdx.x * 16;
  const int h0   = blockIdx.y * 16;
  const size_t base = (size_t)blockIdx.z * TT * CC;

  const int mrow = lane & 15;
  const int kofs = (lane >> 4) * 8;

  // Q as A operand, pre-scaled by 1/sqrt(D)=0.25; padded dims 16..31 are zero
  v16h qa = {};
  {
    const float* qp = Q + base + (size_t)(q0 + mrow) * CC + h0 + kofs;
#pragma unroll
    for (int j = 0; j < 8; ++j) qa[j] = (half_t)(qp[j] * 0.25f);
  }

  v8f acc = {};
  float mi[8], li[8];
#pragma unroll
  for (int r = 0; r < 8; ++r) { mi[r] = -3.0e38f; li[r] = 0.0f; }

  for (int kt = 0; kt < TT / 32; ++kt) {
    const int key0 = kt * 32;

    // V chunk directly as B operand: lane = head dim column, 16 keys in table order
    v16h vb;
    {
      const float* vp = V + base + (size_t)(key0 + kofs) * CC + h0 + (lane & 15);
#pragma unroll
      for (int j = 0; j < 16; ++j) vb[j] = (half_t)vp[(size_t)(j < 8 ? j : j + 8) * CC];
    }
    // K chunk to LDS (row-major), read back transposed for the B operand
    {
      const float* kp = Kin + base + (size_t)(key0 + lane) * CC + h0;
#pragma unroll
      for (int j = 0; j < 16; ++j) Ks[lane * 16 + j] = (half_t)kp[j];
    }
    __syncthreads();

    v16h b0 = {}, b1 = {};
    {                              // elements 8..15 (dims 16..31) are padding (zero)
#pragma unroll
      for (int n = 0; n < 8; ++n) {
        b0[n] = Ks[(lane & 15) * 16 + kofs + n];         // key = key0+(lane&15), dim = kofs+n
        b1[n] = Ks[((lane & 15) + 16) * 16 + kofs + n];  // keys key0+16..key0+31
      }
    }
    v8f s0 = {}, s1 = {};
    s0 = WMMA_F16(qa, b0, s0);
    s1 = WMMA_F16(qa, b1, s1);

    // online softmax; row = (lane>=16 ? r+8 : r), 16 lanes per row half
#pragma unroll
    for (int r = 0; r < 8; ++r) {
      float t0 = s0[r], t1 = s1[r];
      float tm = fmaxf(t0, t1);
#pragma unroll
      for (int off = 1; off < 16; off <<= 1) tm = fmaxf(tm, __shfl_xor(tm, off, 32));
      float mnew  = fmaxf(mi[r], tm);
      float alpha = __expf(mi[r] - mnew);
      float p0 = __expf(t0 - mnew);
      float p1 = __expf(t1 - mnew);
      float rs = p0 + p1;
#pragma unroll
      for (int off = 1; off < 16; off <<= 1) rs += __shfl_xor(rs, off, 32);
      li[r] = li[r] * alpha + rs;
      mi[r] = mnew;
      acc[r] *= alpha;
      int mm = (lane >> 4) * 8 + r;
      int n  = lane & 15;
      Ps[mm * 32 + n]      = (half_t)p0;
      Ps[mm * 32 + n + 16] = (half_t)p1;
    }
    __syncthreads();

    // P as A operand (K = 32 keys), V already in B layout
    const half_t* pp = &Ps[mrow * 32];
    v8h plo = *reinterpret_cast<const v8h*>(pp + kofs);
    v8h phi = *reinterpret_cast<const v8h*>(pp + kofs + 16);
    acc = WMMA_F16(cat8(plo, phi), vb, acc);
    __syncthreads();
  }

#pragma unroll
  for (int r = 0; r < 8; ++r) {
    int mm = (lane >> 4) * 8 + r;
    Y[base + (size_t)(q0 + mm) * CC + h0 + (lane & 15)] = acc[r] / li[r];
  }
}

// ---------------------------------------------------------------------------
// LayerNorm (one wave per token). Optional affine (g,b) and/or AdaLN.
// ---------------------------------------------------------------------------
__global__ __launch_bounds__(256) void ln_kernel(
    const float* __restrict__ X, const float* __restrict__ g,
    const float* __restrict__ bln, const float* __restrict__ ada,
    float* __restrict__ Y)
{
  const int lane = threadIdx.x & 31;
  const int wave = threadIdx.x >> 5;
  const int tok  = blockIdx.x * 8 + wave;          // grid = BT/8
  const float* xp = X + (size_t)tok * CC;

  float v[8];
  float s = 0.0f;
#pragma unroll
  for (int j = 0; j < 8; ++j) { v[j] = xp[lane + j * 32]; s += v[j]; }
#pragma unroll
  for (int off = 1; off < 32; off <<= 1) s += __shfl_xor(s, off, 32);
  float mean = s * (1.0f / CC);

  float var = 0.0f;
#pragma unroll
  for (int j = 0; j < 8; ++j) { float d = v[j] - mean; var += d * d; }
#pragma unroll
  for (int off = 1; off < 32; off <<= 1) var += __shfl_xor(var, off, 32);
  float rstd = rsqrtf(var * (1.0f / CC) + 1e-5f);

  const int bi = tok >> 11;                        // tok / T
#pragma unroll
  for (int j = 0; j < 8; ++j) {
    int c = lane + j * 32;
    float o = (v[j] - mean) * rstd;
    if (g)   o = o * g[c] + bln[c];
    if (ada) {
      float sc = ada[bi * 512 + c];
      float sh = ada[bi * 512 + 256 + c];
      o = o * (1.0f + sc) + sh;
    }
    Y[(size_t)tok * CC + c] = o;
  }
}

// ---------------------------------------------------------------------------
// AdaLN vector: e = emb[t]; ada[b,0:512] = silu(e) @ W[256,512] + bias
// ---------------------------------------------------------------------------
__global__ __launch_bounds__(256) void adaln_vec_kernel(
    const float* __restrict__ emb, const float* __restrict__ W,
    const float* __restrict__ bias, const int* __restrict__ tstep,
    float* __restrict__ ada)
{
  __shared__ float se[256];
  const int b = blockIdx.x, tid = threadIdx.x;
  const int t = tstep[b];
  float e = emb[t * 256 + tid];
  se[tid] = e / (1.0f + __expf(-e));
  __syncthreads();
#pragma unroll
  for (int hh = 0; hh < 2; ++hh) {
    int j = tid + hh * 256;
    float s = bias[j];
    for (int k = 0; k < 256; ++k) s += se[k] * W[k * 512 + j];
    ada[b * 512 + j] = s;
  }
}

// ---------------------------------------------------------------------------
// Host-side orchestration
// ---------------------------------------------------------------------------
extern "C" void kernel_launch(void* const* d_in, const int* in_sizes, int n_in,
                              void* d_out, int out_size, void* d_ws, size_t ws_size,
                              hipStream_t stream) {
  (void)in_sizes; (void)n_in; (void)out_size; (void)ws_size;
  auto F = [&](int i) { return (const float*)d_in[i]; };

  // Input layout (setup_inputs dict order, params flattened depth-first):
  const float* x    = F(0);
  const float* img  = F(1);
  const float* plab = F(2);
  const int*   tst  = (const int*)d_in[3];
  // 4: ln1.g 5: ln1.b 6: ln3.g 7: ln3.b
  // 8-10: ln2{emb,lw,lb} 11-13: ln2_1 14-16: ln2_2
  // 17-24: attn1{kw,kb,qw,qb,vw,vb,pw,pb} 25-32: attn2 33-40: attn3 41-48: attn4
  // 49: fc1.w 50: fc1.b 51: fc2.w 52: fc2.b

  // ---- f32 arena ----
  float* ws   = (float*)d_ws;
  float* hbuf = ws + 0 * (size_t)BTC;    // adaln/ln scratch; also plab_ln
  float* qbuf = ws + 1 * (size_t)BTC;
  float* kbuf = ws + 2 * (size_t)BTC;
  float* vbuf = ws + 3 * (size_t)BTC;
  float* ybuf = ws + 4 * (size_t)BTC;
  float* xcur = ws + 5 * (size_t)BTC;
  float* adab = ws + 6 * (size_t)BTC;    // 3 * B * 512 floats
  float* hidf = qbuf;                    // 16MB contiguous reuse for MLP hidden (f32)

  // ---- f16 arena (after adab, 16KB pad keeps 16B alignment) ----
  half_t* hb   = (half_t*)(ws + 6 * (size_t)BTC + 4096);
  half_t* xh_a = hb;                          // qk-input mirror
  half_t* xh_b = xh_a + (size_t)BTC;          // v-input mirror
  half_t* yh   = xh_b + (size_t)BTC;          // attn-out mirror
  half_t* hidh = yh   + (size_t)BTC;          // MLP hidden mirror (BT*MLPH)
  half_t* wt   = hidh + (size_t)BT * MLPH;    // weight arena (f16)

  float* out_x = (float*)d_out;
  float* plab2 = out_x + (size_t)BTC;

  auto cvt = [&](const float* s, half_t* d, int n) {
    cvt_kernel<<<n / 1024, 256, 0, stream>>>(s, d, n);
  };

  // Convert all GEMM weights to f16 once per call.
  half_t* WH[64] = {};
  {
    size_t off = 0;
    const int widx[18] = {17,19,21,23, 25,27,29,31, 33,35,37,39, 41,43,45,47, 49,51};
    for (int i = 0; i < 18; ++i) {
      int n = (i < 16) ? CC * CC : CC * MLPH;   // fc1 & fc2 are both 256*1024 elems
      WH[widx[i]] = wt + off;
      cvt(F(widx[i]), wt + off, n);
      off += n;
    }
  }

  auto gemm = [&](const half_t* Xh, const half_t* Wh, const float* bias,
                  const float* R, float* Y, int M, int N, int K, int act) {
    dim3 grid(N / 64, M / 64);
    gemm16_kernel<<<grid, 128, 0, stream>>>(Xh, Wh, bias, R, Y, M, N, K, act);
  };
  auto attn = [&](int p, const float* qkin, const float* vin,
                  const float* resid, float* outp) {
    cvt(qkin, xh_a, BTC);
    cvt(vin,  xh_b, BTC);
    gemm(xh_a, WH[p + 2], F(p + 3), nullptr, qbuf, BT, CC, CC, 0);   // query
    gemm(xh_a, WH[p + 0], F(p + 1), nullptr, kbuf, BT, CC, CC, 0);   // key
    gemm(xh_b, WH[p + 4], F(p + 5), nullptr, vbuf, BT, CC, CC, 0);   // value
    flash_kernel<<<dim3(TT / 16, HH, BB), 32, 0, stream>>>(qbuf, kbuf, vbuf, ybuf);
    cvt(ybuf, yh, BTC);
    gemm(yh, WH[p + 6], F(p + 7), resid, outp, BT, CC, CC, 0);       // proj + residual
  };
  auto adaln = [&](int p, const float* Xin, float* ada, float* Yout) {
    adaln_vec_kernel<<<BB, 256, 0, stream>>>(F(p + 0), F(p + 1), F(p + 2), tst, ada);
    ln_kernel<<<BT / 8, 256, 0, stream>>>(Xin, nullptr, nullptr, ada, Yout);
  };

  // 1) plab_ln = LN(plab; ln1)
  ln_kernel<<<BT / 8, 256, 0, stream>>>(plab, F(4), F(5), nullptr, hbuf);
  // 2) plab2 = plab + attn1(qk=img, v=plab_ln)   -> written to output slot
  attn(17, img, hbuf, plab, plab2);
  // 3) h = AdaLN(ln2, x, t);  x1 = x + attn2(h, h)
  adaln(8, x, adab + 0 * BB * 512, hbuf);
  attn(25, hbuf, hbuf, x, xcur);
  // 4) h = AdaLN(ln2_1, x1);  x2 = x1 + attn3(qk=plab2, v=h)
  adaln(11, xcur, adab + 1 * BB * 512, hbuf);
  attn(33, plab2, hbuf, xcur, xcur);
  // 5) h = AdaLN(ln2_2, x2);  x3 = x2 + attn4(qk=img, v=h)
  adaln(14, xcur, adab + 2 * BB * 512, hbuf);
  attn(41, img, hbuf, xcur, xcur);
  // 6) MLP: out_x = x3 + fc2(gelu(fc1(LN(x3; ln3))))
  ln_kernel<<<BT / 8, 256, 0, stream>>>(xcur, F(6), F(7), nullptr, hbuf);
  cvt(hbuf, xh_a, BTC);
  gemm(xh_a, WH[49], F(50), nullptr, hidf, BT, MLPH, CC, 1);         // fc1 + GELU
  cvt(hidf, hidh, BT * MLPH);
  gemm(hidh, WH[51], F(52), xcur, out_x, BT, CC, MLPH, 0);           // fc2 + residual
}
